// ArcFaceLoss_76716705841965
// MI455X (gfx1250) — matrix operand order repaired and checked by reading, not compiled, not run on hardware
//
#include <hip/hip_runtime.h>
#include <math.h>

#define N_ROWS 2048
#define DIMK   128
#define NCLS   100000
#define ANGLE  0.5f
#define BM     128
#define BN     160   // 160 * 625 == 100000 exactly -> no bounds checks, EXEC all-ones
#define CSTR   24    // halves per column entry in swizzled B tile (16 data + 8 pad)

typedef __attribute__((ext_vector_type(16))) _Float16 v16h;
typedef __attribute__((ext_vector_type(8)))  _Float16 v8h;
typedef __attribute__((ext_vector_type(2)))  _Float16 v2h;
typedef __attribute__((ext_vector_type(8)))  float    v8f;
typedef __attribute__((ext_vector_type(4)))  int      v4i;

#if __has_builtin(__builtin_amdgcn_global_load_async_to_lds_b128)
typedef __attribute__((address_space(1))) v4i* gas_v4i_p;   // global int4*
typedef __attribute__((address_space(3))) v4i* lds_v4i_p;   // LDS int4*
#endif

// ---------------- Kernel 1: normalize feature rows, zero row-sum accumulator ----
__global__ void __launch_bounds__(128)
prep_features(const float* __restrict__ feat,
              float* __restrict__ fhat32,
              _Float16* __restrict__ fhat16,
              float* __restrict__ down) {
  const int row = blockIdx.x;
  const int t   = threadIdx.x;                 // 128 threads = 4 waves (wave32)
  float v  = feat[row * DIMK + t];
  float ss = v * v;
#pragma unroll
  for (int m = 16; m >= 1; m >>= 1) ss += __shfl_xor(ss, m);
  __shared__ float part[4];
  if ((t & 31) == 0) part[t >> 5] = ss;
  __syncthreads();
  const float inv = rsqrtf(part[0] + part[1] + part[2] + part[3]);
  const float fh  = v * inv;
  fhat32[row * DIMK + t] = fh;
  fhat16[row * DIMK + t] = (_Float16)fh;
  if (t == 0) down[row] = 0.0f;                // re-zeroed every launch (graph replay safe)
}

// ---------------- Kernel 2: inverse column norms of w ---------------------------
__global__ void __launch_bounds__(256)
wnorm_kernel(const float* __restrict__ w, float* __restrict__ inv_wn) {
  const int c = blockIdx.x * 256 + threadIdx.x;
  if (c >= NCLS) return;
  float s = 0.0f;
#pragma unroll 8
  for (int d = 0; d < DIMK; ++d) {
    if ((d & 31) == 0 && d + 32 < DIMK)        // stream ahead -> global_prefetch_b8
      __builtin_prefetch(w + (size_t)(d + 32) * NCLS + c, 0, 1);
    const float x = w[(size_t)d * NCLS + c];   // coalesced across c
    s += x * x;
  }
  inv_wn[c] = rsqrtf(s);
}

// ---------------- Kernel 3: target-column cosines -------------------------------
__global__ void __launch_bounds__(128)
target_logit(const float* __restrict__ fhat32, const float* __restrict__ w,
             const int* __restrict__ target, const float* __restrict__ inv_wn,
             float* __restrict__ ct) {
  const int row = blockIdx.x;
  const int t   = threadIdx.x;                 // 128 threads
  const int cls = target[row];
  float x = fhat32[row * DIMK + t] * w[(size_t)t * NCLS + cls];
#pragma unroll
  for (int m = 16; m >= 1; m >>= 1) x += __shfl_xor(x, m);
  __shared__ float part[4];
  if ((t & 31) == 0) part[t >> 5] = x;
  __syncthreads();
  if (t == 0) ct[row] = (part[0] + part[1] + part[2] + part[3]) * inv_wn[cls];
}

// ---------------- Kernel 4: fused WMMA GEMM + exp + row-sum ---------------------
// Grid (625, 16); block 256 threads = 8 waves. Wave w owns rows [mb+16w, mb+16w+16)
// and all 10 column fragments of the BN=160 strip. K=128 = 4 x wmma_f32_16x16x32_f16.
//
// B tile is stored fragment-contiguous in LDS: element (k, c) lives at
//   Bs[((k/16)*BN + c)*CSTR + (k%16)]
// so a 32x16 B fragment is two ds_load_b128 per lane. CSTR=24 halves (48 B)
// puts lane l's b128 at dword-bank 12*l mod 64 -> all 16 starts distinct,
// full 64-bank utilization.
__global__ void __launch_bounds__(256)
gemm_exp_rowsum(const _Float16* __restrict__ fhat16,   // [2048][128] row-major f16
                const float*    __restrict__ w,        // [128][100000] row-major f32
                const float*    __restrict__ inv_wn,   // [100000]
                float*          __restrict__ down) {   // [2048] f32 accumulators
  __shared__ _Float16 As[BM * DIMK];                   // 32 KB
  __shared__ _Float16 Bs[(DIMK / 16) * BN * CSTR];     // 60 KB  (of 320 KB WGP LDS)

  const int tid = threadIdx.x;
  const int cb  = blockIdx.x * BN;     // column base
  const int mb  = blockIdx.y * BM;     // row base

  // ---- stage A tile: straight f16 copy, async global->LDS when available ----
  {
    const _Float16* src = fhat16 + (size_t)mb * DIMK;
#if __has_builtin(__builtin_amdgcn_global_load_async_to_lds_b128)
    for (int i = tid; i < (BM * DIMK) / 8; i += 256) {
      __builtin_amdgcn_global_load_async_to_lds_b128(
          (gas_v4i_p)(src + i * 8), (lds_v4i_p)(As + i * 8), 0, 0);
    }
#if __has_builtin(__builtin_amdgcn_s_wait_asynccnt)
    __builtin_amdgcn_s_wait_asynccnt(0);
#else
    asm volatile("s_wait_asynccnt 0x0" ::: "memory");
#endif
#else
    const v8h* vsrc = (const v8h*)src;
    v8h*       vdst = (v8h*)As;
    for (int i = tid; i < (BM * DIMK) / 8; i += 256) vdst[i] = vsrc[i];
#endif
  }
  // ---- stage B tile: f32 global -> f16 swizzled LDS, k-pair per thread-iter ----
  {
    for (int p = tid; p < (DIMK / 2) * BN; p += 256) {   // 64 k-pairs x 160 cols
      const int kp = p / BN;          // k-pair index
      const int c  = p % BN;          // coalesced across threads
      const int k  = kp * 2;
      const float x0 = w[(size_t)k * NCLS + cb + c];
      const float x1 = w[(size_t)(k + 1) * NCLS + cb + c];
      v2h h; h[0] = (_Float16)x0; h[1] = (_Float16)x1;
      *(v2h*)(&Bs[((k >> 4) * BN + c) * CSTR + (k & 15)]) = h;
    }
  }
  __syncthreads();

  const int wave = tid >> 5;
  const int lane = tid & 31;
  const int l    = lane & 15;          // fragment column / A row select
  const int hi   = lane >> 4;          // lane half (K-split per ISA 7.12.2)

  v8f acc[10];
#pragma unroll
  for (int nf = 0; nf < 10; ++nf) acc[nf] = {};

  const _Float16* arow = As + (wave * 16 + l) * DIMK;
#pragma unroll
  for (int ks = 0; ks < 4; ++ks) {
    const int kk = ks * 32;
    // A 16x32 f16 fragment: lanes 0-15 hold K {kk..kk+15},
    //                       lanes 16-31 hold K {kk+16..kk+31}
    const v8h alo = *(const v8h*)(arow + kk + hi * 16);
    const v8h ahi = *(const v8h*)(arow + kk + hi * 16 + 8);
    const v16h a = __builtin_shufflevector(alo, ahi,
        0, 1, 2, 3, 4, 5, 6, 7, 8, 9, 10, 11, 12, 13, 14, 15);

#pragma unroll
    for (int nf = 0; nf < 10; ++nf) {
      // halves j=0..15 are K = kk + hi*16 + j for column nf*16+l: contiguous now
      const _Float16* bp = Bs + (((ks * 2 + hi) * BN) + nf * 16 + l) * CSTR;
      const v8h blo = *(const v8h*)(bp);
      const v8h bhi = *(const v8h*)(bp + 8);
      const v16h b = __builtin_shufflevector(blo, bhi,
          0, 1, 2, 3, 4, 5, 6, 7, 8, 9, 10, 11, 12, 13, 14, 15);
      acc[nf] = __builtin_amdgcn_wmma_f32_16x16x32_f16(
          false, a, false, b, (short)0, acc[nf], false, false);
    }
  }

  // ---- epilogue: scale by inv||w_col||, exp, per-row partial sums ----
  float rs[8];
#pragma unroll
  for (int v = 0; v < 8; ++v) rs[v] = 0.0f;
#pragma unroll
  for (int nf = 0; nf < 10; ++nf) {
    const float s = inv_wn[cb + nf * 16 + l];
#pragma unroll
    for (int v = 0; v < 8; ++v) rs[v] += __expf(acc[nf][v] * s);
  }
  // reduce across the 16 lanes of each half (C/D layout: VGPR v = row hi*8+v)
#pragma unroll
  for (int m = 1; m < 16; m <<= 1) {
#pragma unroll
    for (int v = 0; v < 8; ++v) rs[v] += __shfl_xor(rs[v], m);
  }
  if (l == 0) {
    const int rowbase = mb + wave * 16 + hi * 8;
#pragma unroll
    for (int v = 0; v < 8; ++v) atomicAdd(&down[rowbase + v], rs[v]);
  }
}

// ---------------- Kernel 5: final scalar loss -----------------------------------
__global__ void __launch_bounds__(256)
final_loss(const float* __restrict__ ct, const float* __restrict__ down,
           float* __restrict__ out) {
  __shared__ float part[256];
  const float cA = cosf(ANGLE), sA = sinf(ANGLE);
  float s = 0.0f;
  for (int i = threadIdx.x; i < N_ROWS; i += 256) {
    const float c   = ct[i];
    const float sn  = sqrtf(fmaxf(0.0f, 1.0f - c * c));
    const float ctp = c * cA - sn * sA;          // cos(theta + ANGLE)
    const float top = __expf(ctp);
    const float den = down[i] - __expf(c) + top; // swap target term
    s += logf(den) - ctp;                        // -log(top/den)
  }
  part[threadIdx.x] = s;
  __syncthreads();
  for (int st = 128; st > 0; st >>= 1) {
    if (threadIdx.x < st) part[threadIdx.x] += part[threadIdx.x + st];
    __syncthreads();
  }
  if (threadIdx.x == 0) out[0] = part[0] / (float)N_ROWS;
}

// ---------------- launcher ------------------------------------------------------
extern "C" void kernel_launch(void* const* d_in, const int* in_sizes, int n_in,
                              void* d_out, int out_size, void* d_ws, size_t ws_size,
                              hipStream_t stream) {
  const float* feat   = (const float*)d_in[0];
  const int*   target = (const int*)d_in[1];
  const float* w      = (const float*)d_in[2];

  char* ws = (char*)d_ws;
  float*    fhat32 = (float*)(ws);                               // 1,048,576 B
  _Float16* fhat16 = (_Float16*)(ws + 1048576);                  //   524,288 B
  float*    inv_wn = (float*)(ws + 1048576 + 524288);            //   400,000 B (pad->401408)
  float*    down   = (float*)(ws + 1048576 + 524288 + 401408);   //     8,192 B
  float*    ctbuf  = (float*)(ws + 1048576 + 524288 + 401408 + 8192); // 8,192 B

  prep_features  <<<N_ROWS, 128, 0, stream>>>(feat, fhat32, fhat16, down);
  wnorm_kernel   <<<(NCLS + 255) / 256, 256, 0, stream>>>(w, inv_wn);
  target_logit   <<<N_ROWS, 128, 0, stream>>>(fhat32, w, target, inv_wn, ctbuf);
  gemm_exp_rowsum<<<dim3(NCLS / BN, N_ROWS / BM), 256, 0, stream>>>(fhat16, w, inv_wn, down);
  final_loss     <<<1, 256, 0, stream>>>(ctbuf, down, (float*)d_out);
}
